// RosaBase_86423331930411
// MI455X (gfx1250) — compile-verified, matrix operand order checked by reading, not run on hardware
//
#include <hip/hip_runtime.h>
#include <hip/hip_bf16.h>

// ---------------------------------------------------------------------------
// RoSA bits-attention for MI455X (gfx1250, wave32, WMMA 16x16x32 f16).
//
// Math: sim[t,j] = 0.5 + (1/16) * sum_b tanh(q[t,b]/2) * tanh(k[j,b]/2)
// score[t,j] = sum_{d<4} sim[t-d,j-d]  ==  Qext[t].Kext[j]/16 + 0.5*min(4,j+1)
//   (per-row softmax constants dropped), Qext/Kext = 32-dim suffix stacks
//   -> one v_wmma_f32_16x16x32_f16 per 16x16 score tile.
// Flash-attention online softmax; P tile goes C-layout -> LDS -> A-layout,
// then one WMMA accumulates P @ sigmoid(v) (v stored transposed [H*8, S]).
// GEMM loops are double-buffered (k-loop unrolled x2, compile-time buffer
// indices) so fragment loads for chunk k+32 overlap WMMAs on chunk k.
// ---------------------------------------------------------------------------

typedef __attribute__((ext_vector_type(16))) _Float16 v16h;
typedef __attribute__((ext_vector_type(8)))  _Float16 h8;
typedef __attribute__((ext_vector_type(8)))  float    v8f;

#define S_LEN 2048
#define HID   1024
#define NH    16
#define DQK   128   // NH*8
#define DV    128

union Frag { h8 h[2]; v16h v; };

__device__ __forceinline__ h8 ld8(const _Float16* p) { return *(const h8*)p; }

__device__ __forceinline__ v8f wmma_f16(v16h a, v16h b, v8f c) {
  return __builtin_amdgcn_wmma_f32_16x16x32_f16(false, a, false, b, (short)0, c,
                                                false, false);
}

__device__ __forceinline__ float rmax_h(float v) {
  v = fmaxf(v, __shfl_xor(v, 1, 32));
  v = fmaxf(v, __shfl_xor(v, 2, 32));
  v = fmaxf(v, __shfl_xor(v, 4, 32));
  v = fmaxf(v, __shfl_xor(v, 8, 32));
  return v;   // reduces across 16-lane half (same M rows), xor<16 stays in half
}
__device__ __forceinline__ float rsum_h(float v) {
  v += __shfl_xor(v, 1, 32);
  v += __shfl_xor(v, 2, 32);
  v += __shfl_xor(v, 4, 32);
  v += __shfl_xor(v, 8, 32);
  return v;
}

// ---------------- f32 -> f16 convert ----------------
__global__ void rosa_cvt(const float* __restrict__ s, _Float16* __restrict__ d,
                         int n) {
  int i = blockIdx.x * blockDim.x + threadIdx.x;
  if (i < n) d[i] = (_Float16)s[i];
}

// ---------------- QKV projection + nonlinearity ----------------
// grid (S/128, 6), 128 threads (4 waves, each wave: 32 rows x 64 cols).
// by: 0-1 -> q (tanh(x/2)), 2-3 -> k, 4-5 -> v (sigmoid, stored transposed)
__global__ __launch_bounds__(128)
void rosa_proj(const _Float16* __restrict__ A, const _Float16* __restrict__ qw,
               const _Float16* __restrict__ kw, const _Float16* __restrict__ vw,
               _Float16* __restrict__ qp, _Float16* __restrict__ kp,
               _Float16* __restrict__ vbT) {
  const int t0b = blockIdx.x * 128;
  const int by = blockIdx.y;
  const int wsel = by >> 1;
  const int c0 = (by & 1) * 64;
  const _Float16* W = (wsel == 0) ? qw : (wsel == 1) ? kw : vw;
  const int wave = threadIdx.x >> 5, lane = threadIdx.x & 31;
  const int hh = lane >> 4, ln = lane & 15;
  const int t0w = t0b + wave * 32;

  v8f acc[2][4];
  {
    v8f z = {};
#pragma unroll
    for (int r = 0; r < 2; ++r)
#pragma unroll
      for (int nt = 0; nt < 4; ++nt) acc[r][nt] = z;
  }

  Frag a[2][2], b[2][4];
  auto loadK = [&](int buf, int k0) {
#pragma unroll
    for (int r = 0; r < 2; ++r) {
      const _Float16* ar = A + (size_t)(t0w + r * 16 + ln) * HID + k0;
      a[buf][r].h[0] = ld8(ar + 8 * hh);
      a[buf][r].h[1] = ld8(ar + 16 + 8 * hh);
    }
#pragma unroll
    for (int nt = 0; nt < 4; ++nt) {
      const _Float16* br = W + (size_t)(c0 + nt * 16 + ln) * HID + k0;
      b[buf][nt].h[0] = ld8(br + 8 * hh);
      b[buf][nt].h[1] = ld8(br + 16 + 8 * hh);
    }
  };
  auto mma = [&](int buf) {
#pragma unroll
    for (int r = 0; r < 2; ++r)
#pragma unroll
      for (int nt = 0; nt < 4; ++nt)
        acc[r][nt] = wmma_f16(a[buf][r].v, b[buf][nt].v, acc[r][nt]);
  };

  loadK(0, 0);
  for (int k0 = 0; k0 < HID; k0 += 64) {   // x2 unrolled double buffer
    loadK(1, k0 + 32);                      // overlaps WMMAs on buf 0
    mma(0);
    if (k0 + 64 < HID) loadK(0, k0 + 64);   // overlaps WMMAs on buf 1
    mma(1);
  }

#pragma unroll
  for (int r = 0; r < 2; ++r)
#pragma unroll
    for (int nt = 0; nt < 4; ++nt)
#pragma unroll
      for (int e = 0; e < 8; ++e) {
        int t = t0w + r * 16 + e + 8 * hh;  // C layout: half0 rows 0-7, half1 8-15
        int c = c0 + nt * 16 + ln;
        float x = acc[r][nt][e];
        if (wsel == 0)      qp[(size_t)t * DQK + c] = (_Float16)tanhf(0.5f * x);
        else if (wsel == 1) kp[(size_t)t * DQK + c] = (_Float16)tanhf(0.5f * x);
        else                vbT[(size_t)c * S_LEN + t] =
                                (_Float16)(1.0f / (1.0f + __expf(-x)));
      }
}

// ---------------- flash attention over suffix-stacked bits ----------------
// grid (S/64, NH), 128 threads; wave -> 16 query rows, j in chunks of 32.
__global__ __launch_bounds__(128)
void rosa_attn(const _Float16* __restrict__ qp, const _Float16* __restrict__ kp,
               const _Float16* __restrict__ vbT, _Float16* __restrict__ outb) {
  __shared__ _Float16 plds[4][16][40];   // per-wave P tile (16x32, padded)
  const int h = blockIdx.y;
  const int t0b = blockIdx.x * 64;
  const int wave = threadIdx.x >> 5, lane = threadIdx.x & 31;
  const int hh = lane >> 4, ln = lane & 15;
  const int t0w = t0b + wave * 16;

  // Qext fragment: K-dim = [d0 bits | d1 | d2 | d3], zero pad t-d < 0.
  Frag aq;
  {
    h8 z = {};
    int r = t0w + ln;
    aq.h[0] = (r - hh >= 0) ? ld8(qp + (size_t)(r - hh) * DQK + h * 8) : z;
    aq.h[1] = (r - hh - 2 >= 0) ? ld8(qp + (size_t)(r - hh - 2) * DQK + h * 8) : z;
  }

  v8f oacc = {};
  float mrun[8], lrun[8];
#pragma unroll
  for (int e = 0; e < 8; ++e) { mrun[e] = -1e30f; lrun[e] = 0.0f; }

  const int jend = t0b + 63;
  for (int j0 = 0; j0 <= jend; j0 += 32) {
    if (j0 > t0w + 15) break;  // wave-uniform; no cross-wave sync in this loop

    if (j0 + 32 <= jend) {
      __builtin_prefetch((const void*)(kp + (size_t)(j0 + 32) * DQK + h * 8), 0, 1);
      __builtin_prefetch((const void*)(vbT + (size_t)h * 8 * S_LEN + j0 + 32), 0, 1);
    }

    // Issue all global fragment loads up-front (K~ tiles + V tile) so they
    // overlap the score WMMAs and the softmax VALU block below.
    Frag bk0, bk1, bv;
    {
      h8 z = {};
      int c = j0 + ln;
      bk0.h[0] = (c - hh >= 0) ? ld8(kp + (size_t)(c - hh) * DQK + h * 8) : z;
      bk0.h[1] = (c - hh - 2 >= 0) ? ld8(kp + (size_t)(c - hh - 2) * DQK + h * 8) : z;
      c = j0 + 16 + ln;  // c >= 16, suffix always in range
      bk1.h[0] = ld8(kp + (size_t)(c - hh) * DQK + h * 8);
      bk1.h[1] = ld8(kp + (size_t)(c - hh - 2) * DQK + h * 8);
      if (ln < 8) {
        const _Float16* vb = vbT + (size_t)(h * 8 + ln) * S_LEN;
        bv.h[0] = ld8(vb + j0 + 8 * hh);
        bv.h[1] = ld8(vb + j0 + 16 + 8 * hh);
      } else { bv.h[0] = z; bv.h[1] = z; }
    }

    v8f zc = {};
    v8f s0 = wmma_f16(aq.v, bk0.v, zc);
    v8f s1 = wmma_f16(aq.v, bk1.v, zc);

    const int colA = j0 + ln, colB = j0 + 16 + ln;
    const float cbA = 0.5f * fminf(4.0f, (float)(colA + 1));
    const float cbB = 0.5f * fminf(4.0f, (float)(colB + 1));
#pragma unroll
    for (int e = 0; e < 8; ++e) {
      const int trow = t0w + e + 8 * hh;
      float la = s0[e] * 0.0625f + cbA;  if (colA > trow) la = -1e30f;
      float lb = s1[e] * 0.0625f + cbB;  if (colB > trow) lb = -1e30f;
      float m = rmax_h(fmaxf(la, lb));
      float mn = fmaxf(mrun[e], m);
      float alpha = __expf(mrun[e] - mn);
      float pa = __expf(la - mn);
      float pb = __expf(lb - mn);
      s0[e] = pa; s1[e] = pb;
      lrun[e] = lrun[e] * alpha + rsum_h(pa + pb);
      mrun[e] = mn;
      oacc[e] *= alpha;
    }

    // C-layout P -> LDS -> A-layout P (per-wave region, same-wave RAW)
#pragma unroll
    for (int e = 0; e < 8; ++e) {
      plds[wave][e + 8 * hh][ln]      = (_Float16)s0[e];
      plds[wave][e + 8 * hh][ln + 16] = (_Float16)s1[e];
    }
    asm volatile("s_wait_dscnt 0" ::: "memory");
    Frag ap;
    ap.h[0] = ld8(&plds[wave][ln][8 * hh]);
    ap.h[1] = ld8(&plds[wave][ln][16 + 8 * hh]);

    oacc = wmma_f16(ap.v, bv.v, oacc);
  }

  if (ln < 8) {
#pragma unroll
    for (int e = 0; e < 8; ++e) {
      int t = t0w + e + 8 * hh;
      outb[(size_t)t * DV + h * 8 + ln] = (_Float16)(oacc[e] / lrun[e]);
    }
  }
}

// ---------------- binary-embedding interpolation ----------------
__global__ void rosa_interp(const _Float16* __restrict__ outb,
                            const float* __restrict__ e0,
                            const float* __restrict__ e1,
                            _Float16* __restrict__ oemb) {
  int i = blockIdx.x * blockDim.x + threadIdx.x;
  if (i < S_LEN * DV) {
    int c = i & (DV - 1);
    float b = (float)outb[i];
    oemb[i] = (_Float16)(e0[c] + (e1[c] - e0[c]) * b);
  }
}

// ---------------- fused gate GEMM + output GEMM + blend ----------------
// grid (S/64, HID/64), 128 threads; wave -> 16 rows x 64 cols.
__global__ __launch_bounds__(128)
void rosa_final(const _Float16* __restrict__ A, const _Float16* __restrict__ gw,
                const _Float16* __restrict__ oemb, const _Float16* __restrict__ ow,
                const float* __restrict__ inject, float* __restrict__ out) {
  const int t0b = blockIdx.x * 64;
  const int c0b = blockIdx.y * 64;
  const int wave = threadIdx.x >> 5, lane = threadIdx.x & 31;
  const int hh = lane >> 4, ln = lane & 15;
  const int t0w = t0b + wave * 16;

  // ---- output GEMM: out_emb[16 x 128] @ o_w.T tile, K=128 fully unrolled ----
  v8f acco[4];
  {
    v8f z = {};
#pragma unroll
    for (int nt = 0; nt < 4; ++nt) acco[nt] = z;
#pragma unroll
    for (int k0 = 0; k0 < DV; k0 += 32) {
      Frag ao;
      const _Float16* aor = oemb + (size_t)(t0w + ln) * DV + k0;
      ao.h[0] = ld8(aor + 8 * hh);
      ao.h[1] = ld8(aor + 16 + 8 * hh);
#pragma unroll
      for (int nt = 0; nt < 4; ++nt) {
        Frag bo;
        const _Float16* br = ow + (size_t)(c0b + nt * 16 + ln) * DV + k0;
        bo.h[0] = ld8(br + 8 * hh);
        bo.h[1] = ld8(br + 16 + 8 * hh);
        acco[nt] = wmma_f16(ao.v, bo.v, acco[nt]);
      }
    }
  }

  // ---- gate GEMM: hidden[16 x 1024] @ gate_w.T tile, double buffered ----
  v8f accg[4];
  {
    v8f z = {};
#pragma unroll
    for (int nt = 0; nt < 4; ++nt) accg[nt] = z;
  }
  Frag a[2], b[2][4];
  auto loadK = [&](int buf, int k0) {
    const _Float16* ar = A + (size_t)(t0w + ln) * HID + k0;
    a[buf].h[0] = ld8(ar + 8 * hh);
    a[buf].h[1] = ld8(ar + 16 + 8 * hh);
#pragma unroll
    for (int nt = 0; nt < 4; ++nt) {
      const _Float16* br = gw + (size_t)(c0b + nt * 16 + ln) * HID + k0;
      b[buf][nt].h[0] = ld8(br + 8 * hh);
      b[buf][nt].h[1] = ld8(br + 16 + 8 * hh);
    }
  };
  auto mma = [&](int buf) {
#pragma unroll
    for (int nt = 0; nt < 4; ++nt)
      accg[nt] = wmma_f16(a[buf].v, b[buf][nt].v, accg[nt]);
  };

  loadK(0, 0);
  for (int k0 = 0; k0 < HID; k0 += 64) {
    loadK(1, k0 + 32);
    mma(0);
    if (k0 + 64 < HID) loadK(0, k0 + 64);
    mma(1);
  }

#pragma unroll
  for (int nt = 0; nt < 4; ++nt)
#pragma unroll
    for (int e = 0; e < 8; ++e) {
      int t = t0w + e + 8 * hh;
      int c = c0b + nt * 16 + ln;
      float g = 1.0f / (1.0f + __expf(-accg[nt][e]));
      size_t idx = (size_t)t * HID + c;
      out[idx] = acco[nt][e] * g + inject[idx] * (1.0f - g);
    }
}

// ---------------------------------------------------------------------------
extern "C" void kernel_launch(void* const* d_in, const int* in_sizes, int n_in,
                              void* d_out, int out_size, void* d_ws,
                              size_t ws_size, hipStream_t stream) {
  const float* hs     = (const float*)d_in[0];
  const float* inject = (const float*)d_in[1];
  const float* qw     = (const float*)d_in[2];
  const float* kw     = (const float*)d_in[3];
  const float* vw     = (const float*)d_in[4];
  const float* ow     = (const float*)d_in[5];
  const float* gw     = (const float*)d_in[6];
  const float* e0     = (const float*)d_in[7];
  const float* e1     = (const float*)d_in[8];
  float* out = (float*)d_out;

  _Float16* ws = (_Float16*)d_ws;
  size_t o = 0;
  _Float16* hid_h = ws + o; o += (size_t)S_LEN * HID;     // 2M halves
  _Float16* gw_h  = ws + o; o += (size_t)HID * HID;       // 1M
  _Float16* qw_h  = ws + o; o += (size_t)DQK * HID;
  _Float16* kw_h  = ws + o; o += (size_t)DQK * HID;
  _Float16* vw_h  = ws + o; o += (size_t)DV * HID;
  _Float16* ow_h  = ws + o; o += (size_t)HID * DV;
  _Float16* qp    = ws + o; o += (size_t)S_LEN * DQK;
  _Float16* kp    = ws + o; o += (size_t)S_LEN * DQK;
  _Float16* vbT   = ws + o; o += (size_t)S_LEN * DV;
  _Float16* outb  = ws + o; o += (size_t)S_LEN * DV;
  _Float16* oemb  = ws + o; o += (size_t)S_LEN * DV;

  const int T = 256;
  rosa_cvt<<<(S_LEN * HID + T - 1) / T, T, 0, stream>>>(hs, hid_h, S_LEN * HID);
  rosa_cvt<<<(HID * HID + T - 1) / T, T, 0, stream>>>(gw, gw_h, HID * HID);
  rosa_cvt<<<(DQK * HID + T - 1) / T, T, 0, stream>>>(qw, qw_h, DQK * HID);
  rosa_cvt<<<(DQK * HID + T - 1) / T, T, 0, stream>>>(kw, kw_h, DQK * HID);
  rosa_cvt<<<(DV * HID + T - 1) / T, T, 0, stream>>>(vw, vw_h, DV * HID);
  rosa_cvt<<<(HID * DV + T - 1) / T, T, 0, stream>>>(ow, ow_h, HID * DV);

  rosa_proj<<<dim3(S_LEN / 128, 6), 128, 0, stream>>>(hid_h, qw_h, kw_h, vw_h,
                                                      qp, kp, vbT);
  rosa_attn<<<dim3(S_LEN / 64, NH), 128, 0, stream>>>(qp, kp, vbT, outb);
  rosa_interp<<<(S_LEN * DV + T - 1) / T, T, 0, stream>>>(outb, e0, e1, oemb);
  rosa_final<<<dim3(S_LEN / 64, HID / 64), 128, 0, stream>>>(hid_h, gw_h, oemb,
                                                             ow_h, inject, out);
}
